// MultiHeadAttention_23536420782872
// MI455X (gfx1250) — matrix operand order repaired and checked by reading, not compiled, not run on hardware
//
#include <hip/hip_runtime.h>
#include <hip/hip_bf16.h>
#include <math.h>

// Problem constants (match reference)
#define BB   2
#define LL   2048
#define DD   1024
#define HH   16
#define DKV  64
#define NEGV (-1000000000.0f)

typedef __bf16 bf16;
typedef __attribute__((ext_vector_type(16))) __bf16 v16bf;
typedef __attribute__((ext_vector_type(8)))  float  v8f;
typedef __attribute__((ext_vector_type(4)))  float  v4f;
typedef __attribute__((ext_vector_type(4)))  __bf16 v4bf;

// ---------------------------------------------------------------------------
// WMMA helpers (CDNA5: D = A(16x32 bf16) * B(32x16 bf16) + C(16x16 f32))
// ---------------------------------------------------------------------------
__device__ __forceinline__ v8f wmma_bf16(v16bf a, v16bf b, v8f c) {
  // 8 args: (neg_a, A, neg_b, B, c_mod, C, reuse_a, reuse_b)
  return __builtin_amdgcn_wmma_f32_16x16x32_bf16(
      false, a, false, b, (short)0, c, false, false);
}

// A-operand fragment: lane m holds K {0..15} (lanes 0-15) or
// {16..31} (lanes 16-31) -> two contiguous 8-element (16B) loads.
__device__ __forceinline__ v16bf ld_a_pair(const bf16* p0, const bf16* p1) {
  v16bf r;
#pragma unroll
  for (int i = 0; i < 8; ++i) { r[i] = p0[i]; r[8 + i] = p1[i]; }
  return r;
}

// B-operand fragment: lane holds its column's 16 contiguous K values.
__device__ __forceinline__ v16bf ld_b_16(const bf16* p) {
  v16bf r;
#pragma unroll
  for (int i = 0; i < 16; ++i) r[i] = p[i];
  return r;
}

// Cross-lane reductions within 16-lane half-wave groups (wave32).
// Used only ONCE per wave (softmax merge).
__device__ __forceinline__ float rmax16(float v) {
  v = fmaxf(v, __shfl_xor(v, 1, 32));
  v = fmaxf(v, __shfl_xor(v, 2, 32));
  v = fmaxf(v, __shfl_xor(v, 4, 32));
  v = fmaxf(v, __shfl_xor(v, 8, 32));
  return v;
}
__device__ __forceinline__ float rsum16(float v) {
  v += __shfl_xor(v, 1, 32);
  v += __shfl_xor(v, 2, 32);
  v += __shfl_xor(v, 4, 32);
  v += __shfl_xor(v, 8, 32);
  return v;
}

// ---------------------------------------------------------------------------
// fp32 -> bf16 convert, 4-wide (b128 in / b64 out)
// ---------------------------------------------------------------------------
__global__ void cvt_f32_bf16(const float* __restrict__ src,
                             bf16* __restrict__ dst, int n4) {
  int i = blockIdx.x * blockDim.x + threadIdx.x;
  const int stride = gridDim.x * blockDim.x;
  const v4f* s4 = (const v4f*)src;
  v4bf* d4 = (v4bf*)dst;
  for (; i < n4; i += stride) {
    v4f f = s4[i];
    v4bf o;
    o[0] = (bf16)f[0]; o[1] = (bf16)f[1]; o[2] = (bf16)f[2]; o[3] = (bf16)f[3];
    d4[i] = o;
  }
}

// ---------------------------------------------------------------------------
// Projection GEMM: Y[4096,1024] = X(bf16) @ W^T(bf16) + bias(f32)
// One wave owns a 16(M) x 64(N) tile. 128 WMMAs / wave.
// VMODE 0: dst layout [B,H,L,64] (Q,K) ; VMODE 1: dst layout [B,H,64,L] (V^T)
// ---------------------------------------------------------------------------
template <int VMODE>
__global__ void __launch_bounds__(128)
proj_gemm(const bf16* __restrict__ X, const bf16* __restrict__ W,
          const float* __restrict__ bias, bf16* __restrict__ dst) {
  const int lane = threadIdx.x & 31;
  const int wave = blockIdx.x * (blockDim.x >> 5) + (threadIdx.x >> 5);
  const int m0   = (wave & 255) * 16;   // 256 M tiles
  const int n0   = (wave >> 8) * 64;    // 16 N groups of 64
  const int l15  = lane & 15, half = lane >> 4;

  v8f acc[4] = {};
  const bf16* arow = X + (size_t)(m0 + l15) * DD;
#pragma unroll 1
  for (int k = 0; k < DD; k += 32) {
    __builtin_prefetch(arow + k + 256, 0, 0);  // global_prefetch next tiles
    v16bf a = ld_a_pair(arow + k + half * 16, arow + k + 8 + half * 16);
#pragma unroll
    for (int nt = 0; nt < 4; ++nt) {
      const bf16* bp = W + (size_t)(n0 + nt * 16 + l15) * DD + k + half * 16;
      acc[nt] = wmma_bf16(a, ld_b_16(bp), acc[nt]);
    }
  }
  // epilogue: C/D layout -> slot j is row (j + 8*half), col = lane&15
#pragma unroll
  for (int nt = 0; nt < 4; ++nt) {
    const int n  = n0 + nt * 16 + l15;
    const float bv = bias[n];
    const int h = n >> 6, d = n & 63;
#pragma unroll
    for (int j = 0; j < 8; ++j) {
      const int row = m0 + j + half * 8;  // global M = b*L + seq
      const int b   = row >> 11;          // / L
      const int sq  = row & (LL - 1);
      const float v = acc[nt][j] + bv;
      if (VMODE == 0)
        dst[(((size_t)(b * HH + h)) * LL + sq) * DKV + d] = (bf16)v;
      else
        dst[(((size_t)(b * HH + h)) * DKV + d) * LL + sq] = (bf16)v;
    }
  }
}

// ---------------------------------------------------------------------------
// Fused attention.
// Pass 1: QK^T WMMA + mask; PER-LANE online softmax (no cross-lane ops in the
//         loop); one 16-lane shuffle merge after the loop.
// Pass 2: recompute scores, P -> d_out (nontemporal f32), bf16 P tile staged
//         in LDS to convert C-layout -> A-layout, 4 PV WMMAs per 32-wide step.
// One wave per (b, h, 16-row Q block). 4096 blocks x 32 threads.
// ---------------------------------------------------------------------------
__global__ void __launch_bounds__(32)
attn_kernel(const bf16* __restrict__ qh, const bf16* __restrict__ kh,
            const bf16* __restrict__ vT, const int* __restrict__ mask,
            float* __restrict__ Pout, bf16* __restrict__ ctx) {
  __shared__ bf16 ldsP[16][40];  // 16x32 P tile, padded

  const int lane = threadIdx.x & 31;
  const int l15 = lane & 15, half = lane >> 4, half8 = half * 8;
  const int bid = blockIdx.x;
  const int bh = bid >> 7;      // b*H + h
  const int qt = bid & 127;
  const int b = bh >> 4, h = bh & 15;
  const int q0 = qt * 16;

  const bf16* qbase = qh + (((size_t)bh) * LL + q0) * DKV;
  const bf16* kbase = kh + ((size_t)bh) * LL * DKV;
  const bf16* vbase = vT + ((size_t)bh) * DKV * LL;
  const int*  mbase = mask + (size_t)b * LL * LL + (size_t)(q0 + half8) * LL + l15;
  float*      pbase = Pout + (((size_t)bh) * LL + q0) * LL;

  // Q operand (kept in registers for both passes): two K=32 fragments
  const bf16* qrow = qbase + (size_t)l15 * DKV;
  const v16bf aq0 = ld_a_pair(qrow + half * 16,      qrow + 8 + half * 16);
  const v16bf aq1 = ld_a_pair(qrow + 32 + half * 16, qrow + 40 + half * 16);

  // Per-lane online softmax state (this lane sees cols k0 + (lane&15)).
  float mlane[8], slane[8];
#pragma unroll
  for (int j = 0; j < 8; ++j) { mlane[j] = -3.0e38f; slane[j] = 0.0f; }

  // ---- pass 1: per-lane online stats over all K tiles (no shuffles) ----
#pragma unroll 1
  for (int k0 = 0; k0 < LL; k0 += 16) {
    const bf16* kp = kbase + (size_t)(k0 + l15) * DKV + half * 16;
    __builtin_prefetch(kp + 16 * DKV, 0, 0);
    v8f c = {};
    c = wmma_bf16(aq0, ld_b_16(kp),      c);
    c = wmma_bf16(aq1, ld_b_16(kp + 32), c);
    const int* mp = mbase + k0;
#pragma unroll
    for (int j = 0; j < 8; ++j) {
      float sc = c[j] * 0.125f;                 // 1/sqrt(DK)
      if (mp[(size_t)j * LL] == 0) sc = NEGV;
      const float nm = fmaxf(mlane[j], sc);
      slane[j] = slane[j] * __expf(mlane[j] - nm) + __expf(sc - nm);
      mlane[j] = nm;
    }
  }
  // ---- one-time 16-lane merge: row max + rescaled row sum ----
  float mrow[8], inv[8];
#pragma unroll
  for (int j = 0; j < 8; ++j) {
    const float M = rmax16(mlane[j]);
    const float s = rsum16(slane[j] * __expf(mlane[j] - M));
    mrow[j] = M;
    inv[j]  = 1.0f / s;
  }

  // ---- pass 2: P out + ctx accumulation (32-wide K steps) ----
  v8f cctx[4] = {};
#pragma unroll 1
  for (int k0 = 0; k0 < LL; k0 += 32) {
    v8f c0 = {}, c1 = {};
    {
      const bf16* kp0 = kbase + (size_t)(k0 + l15) * DKV + half * 16;
      __builtin_prefetch(kp0 + 32 * DKV, 0, 0);
      c0 = wmma_bf16(aq0, ld_b_16(kp0),      c0);
      c0 = wmma_bf16(aq1, ld_b_16(kp0 + 32), c0);
      const bf16* kp1 = kbase + (size_t)(k0 + 16 + l15) * DKV + half * 16;
      c1 = wmma_bf16(aq0, ld_b_16(kp1),      c1);
      c1 = wmma_bf16(aq1, ld_b_16(kp1 + 32), c1);
    }
    const int* mp = mbase + k0;
#pragma unroll
    for (int j = 0; j < 8; ++j) {
      float s0 = c0[j] * 0.125f, s1 = c1[j] * 0.125f;
      if (mp[(size_t)j * LL] == 0)      s0 = NEGV;
      if (mp[(size_t)j * LL + 16] == 0) s1 = NEGV;
      const float p0 = __expf(s0 - mrow[j]) * inv[j];
      const float p1 = __expf(s1 - mrow[j]) * inv[j];
      const int row = j + half8;
      // stream attn_weights: dominant 537MB write -> non-temporal
      __builtin_nontemporal_store(p0, pbase + (size_t)row * LL + k0 + l15);
      __builtin_nontemporal_store(p1, pbase + (size_t)row * LL + k0 + 16 + l15);
      ldsP[row][l15]      = (bf16)p0;
      ldsP[row][16 + l15] = (bf16)p1;
    }
    __syncthreads();  // single-wave WG: orders LDS RAW (waits, barrier->nop)
    const v16bf ap = ld_a_pair(&ldsP[l15][half * 16], &ldsP[l15][8 + half * 16]);
#pragma unroll
    for (int nt = 0; nt < 4; ++nt) {
      const bf16* vp = vbase + (size_t)(nt * 16 + l15) * LL + k0 + half * 16;
      cctx[nt] = wmma_bf16(ap, ld_b_16(vp), cctx[nt]);
    }
    __syncthreads();  // protect LDS tile before next iteration overwrites
  }

  // ctx -> [B*L, H*64] bf16 for the dense GEMM
#pragma unroll
  for (int nt = 0; nt < 4; ++nt) {
#pragma unroll
    for (int j = 0; j < 8; ++j) {
      const int row = q0 + j + half8;
      const int col = h * DKV + nt * 16 + l15;
      ctx[((size_t)b * LL + row) * DD + col] = (bf16)cctx[nt][j];
    }
  }
}

// ---------------------------------------------------------------------------
// Dense out GEMM + bias + residual -> fp32 scratch (pre-LayerNorm)
// ---------------------------------------------------------------------------
__global__ void __launch_bounds__(128)
dense_gemm(const bf16* __restrict__ X, const bf16* __restrict__ W,
           const float* __restrict__ bias, const float* __restrict__ resid,
           float* __restrict__ out) {
  const int lane = threadIdx.x & 31;
  const int wave = blockIdx.x * (blockDim.x >> 5) + (threadIdx.x >> 5);
  const int m0 = (wave & 255) * 16;
  const int n0 = (wave >> 8) * 64;
  const int l15 = lane & 15, half = lane >> 4;

  v8f acc[4] = {};
  const bf16* arow = X + (size_t)(m0 + l15) * DD;
#pragma unroll 1
  for (int k = 0; k < DD; k += 32) {
    v16bf a = ld_a_pair(arow + k + half * 16, arow + k + 8 + half * 16);
#pragma unroll
    for (int nt = 0; nt < 4; ++nt) {
      const bf16* bp = W + (size_t)(n0 + nt * 16 + l15) * DD + k + half * 16;
      acc[nt] = wmma_bf16(a, ld_b_16(bp), acc[nt]);
    }
  }
#pragma unroll
  for (int nt = 0; nt < 4; ++nt) {
    const int n = n0 + nt * 16 + l15;
    const float bv = bias[n];
#pragma unroll
    for (int j = 0; j < 8; ++j) {
      const int row = m0 + j + half * 8;
      const size_t idx = (size_t)row * DD + n;
      out[idx] = acc[nt][j] + bv + resid[idx];
    }
  }
}

// ---------------------------------------------------------------------------
// LayerNorm over D=1024, one block (256 thr) per row.
// ---------------------------------------------------------------------------
__global__ void __launch_bounds__(256)
ln_kernel(const float* __restrict__ x, const float* __restrict__ g,
          const float* __restrict__ bta, float* __restrict__ out) {
  __shared__ float sred[16];
  const int row = blockIdx.x;
  const float* xr = x + (size_t)row * DD;
  float v[4];
  float sum = 0.0f, sq = 0.0f;
#pragma unroll
  for (int i = 0; i < 4; ++i) {
    v[i] = xr[threadIdx.x + i * 256];
    sum += v[i];
    sq  += v[i] * v[i];
  }
#pragma unroll
  for (int off = 1; off < 32; off <<= 1) {
    sum += __shfl_xor(sum, off, 32);
    sq  += __shfl_xor(sq,  off, 32);
  }
  const int wid = threadIdx.x >> 5;
  if ((threadIdx.x & 31) == 0) { sred[wid] = sum; sred[8 + wid] = sq; }
  __syncthreads();
  sum = 0.0f; sq = 0.0f;
#pragma unroll
  for (int i = 0; i < 8; ++i) { sum += sred[i]; sq += sred[8 + i]; }
  const float mu  = sum * (1.0f / DD);
  const float var = sq * (1.0f / DD) - mu * mu;
  const float rs  = rsqrtf(var + 1e-6f);
#pragma unroll
  for (int i = 0; i < 4; ++i) {
    const int c = threadIdx.x + i * 256;
    out[(size_t)row * DD + c] = (v[i] - mu) * rs * g[c] + bta[c];
  }
}

// ---------------------------------------------------------------------------
// Host launcher
// ---------------------------------------------------------------------------
extern "C" void kernel_launch(void* const* d_in, const int* in_sizes, int n_in,
                              void* d_out, int out_size, void* d_ws, size_t ws_size,
                              hipStream_t stream) {
  (void)in_sizes; (void)n_in; (void)out_size; (void)ws_size;
  const float* q    = (const float*)d_in[0];
  const float* k    = (const float*)d_in[1];
  const float* v    = (const float*)d_in[2];
  const int*   mask = (const int*)d_in[3];
  const float* wq_w = (const float*)d_in[4];
  const float* wq_b = (const float*)d_in[5];
  const float* wk_w = (const float*)d_in[6];
  const float* wk_b = (const float*)d_in[7];
  const float* wv_w = (const float*)d_in[8];
  const float* wv_b = (const float*)d_in[9];
  const float* dw   = (const float*)d_in[10];
  const float* db   = (const float*)d_in[11];
  const float* lnw  = (const float*)d_in[12];
  const float* lnb  = (const float*)d_in[13];

  const size_t NX = (size_t)BB * LL * DD;  // 4,194,304 activations
  const size_t NW = (size_t)DD * DD;       // 1,048,576 per weight matrix

  char* ws = (char*)d_ws;
  size_t off = 0;
  auto alloc = [&](size_t bytes) -> void* {
    void* p = ws + off;
    off += (bytes + 255) & ~(size_t)255;
    return p;
  };
  bf16* qb  = (bf16*)alloc(NX * 2);
  bf16* kb  = (bf16*)alloc(NX * 2);
  bf16* vb  = (bf16*)alloc(NX * 2);
  bf16* wqb = (bf16*)alloc(NW * 2);
  bf16* wkb = (bf16*)alloc(NW * 2);
  bf16* wvb = (bf16*)alloc(NW * 2);
  bf16* wdb = (bf16*)alloc(NW * 2);
  bf16* qhp = (bf16*)alloc(NX * 2);   // [B,H,L,64]
  bf16* khp = (bf16*)alloc(NX * 2);   // [B,H,L,64]
  bf16* vTp = (bf16*)alloc(NX * 2);   // [B,H,64,L]
  bf16* ctx = (bf16*)alloc(NX * 2);   // [B*L, H*64]
  float* pre_ln = (float*)alloc(NX * 4);

  float* out_f  = (float*)d_out;  // [B,L,D]
  float* attn_f = out_f + NX;     // [B,H,L,L]

  // 1) fp32 -> bf16 (4-wide)
  cvt_f32_bf16<<<1024, 256, 0, stream>>>(q, qb, (int)(NX / 4));
  cvt_f32_bf16<<<1024, 256, 0, stream>>>(k, kb, (int)(NX / 4));
  cvt_f32_bf16<<<1024, 256, 0, stream>>>(v, vb, (int)(NX / 4));
  cvt_f32_bf16<<<512, 256, 0, stream>>>(wq_w, wqb, (int)(NW / 4));
  cvt_f32_bf16<<<512, 256, 0, stream>>>(wk_w, wkb, (int)(NW / 4));
  cvt_f32_bf16<<<512, 256, 0, stream>>>(wv_w, wvb, (int)(NW / 4));
  cvt_f32_bf16<<<512, 256, 0, stream>>>(dw,   wdb, (int)(NW / 4));

  // 2) QKV projections (4096 waves each: 256 M-tiles x 16 N-groups)
  proj_gemm<0><<<1024, 128, 0, stream>>>(qb, wqb, wq_b, qhp);
  proj_gemm<0><<<1024, 128, 0, stream>>>(kb, wkb, wk_b, khp);
  proj_gemm<1><<<1024, 128, 0, stream>>>(vb, wvb, wv_b, vTp);

  // 3) fused attention: (B*H)=32 x (L/16)=128 blocks, 1 wave each
  attn_kernel<<<4096, 32, 0, stream>>>(qhp, khp, vTp, mask, attn_f, ctx);

  // 4) dense + bias + residual
  dense_gemm<<<1024, 128, 0, stream>>>(ctx, wdb, db, q, pre_ln);

  // 5) LayerNorm -> final output
  ln_kernel<<<(int)(BB * LL), 256, 0, stream>>>(pre_ln, lnw, lnb, out_f);
}
